// BiaffineSpanClassifier_12060268167808
// MI455X (gfx1250) — compile-verified, matrix operand order checked by reading, not run on hardware
//
#include <hip/hip_runtime.h>
#include <stddef.h>

// ---------------------------------------------------------------------------
// Problem constants (from reference): B=4, S=1024, I=768, H=64, O=16
// ---------------------------------------------------------------------------
#define BB 4
#define SS 1024
#define II 768
#define HH 64
#define OO 16
#define NEGBIG 1e12f

typedef __attribute__((ext_vector_type(16))) _Float16 v16h;
typedef __attribute__((ext_vector_type(8)))  _Float16 v8h;
typedef __attribute__((ext_vector_type(8)))  float    v8f;

// ---------------------------------------------------------------------------
// Kernel 1: projections + ReLU + RoPE rotation.
//   grid = B*S rows, block = 64 threads (one per hidden dim h).
//   Writes start_rot (f32, for the mid GEMM) and end_rot (f16, WMMA B operand).
// ---------------------------------------------------------------------------
__global__ __launch_bounds__(64) void proj_rope_kernel(
    const float* __restrict__ x,
    const float* __restrict__ Ws, const float* __restrict__ bs,
    const float* __restrict__ We, const float* __restrict__ be,
    float* __restrict__ start_out, _Float16* __restrict__ end_out)
{
    const int row = blockIdx.x;            // 0 .. B*S-1
    const int h   = threadIdx.x;           // 0 .. 63

    __shared__ float xs[II];
    __shared__ float sbuf[HH];
    __shared__ float ebuf[HH];

    const float* xr = x + (size_t)row * II;
    for (int i = h; i < II; i += HH) xs[i] = xr[i];
    __syncthreads();

    float accS = bs[h];
    float accE = be[h];
    for (int i = 0; i < II; ++i) {
        const float xv = xs[i];
        accS = fmaf(xv, Ws[i * HH + h], accS);
        accE = fmaf(xv, We[i * HH + h], accE);
    }
    accS = fmaxf(accS, 0.0f);
    accE = fmaxf(accE, 0.0f);
    sbuf[h] = accS;
    ebuf[h] = accE;
    __syncthreads();

    // RoPE: c_j = cos(p * f[j%32]), s_j = sin(p * f[j%32]),
    // f[m] = 10000^(-2m/64) = exp(-m * ln(10000)/32)
    const int   p  = row & (SS - 1);
    const float fr = __expf(-(float)(h & 31) * 0.28782313662425572f);
    const float ang = (float)p * fr;
    const float c = cosf(ang);
    const float s = sinf(ang);

    // h2[j] = (j even) ? -h[j+1] : h[j-1]
    const float pS = sbuf[h ^ 1];
    const float pE = ebuf[h ^ 1];
    const float h2S = (h & 1) ? pS : -pS;
    const float h2E = (h & 1) ? pE : -pE;

    const float oS = accS * c + h2S * s;
    const float oE = accE * c + h2E * s;

    start_out[(size_t)row * HH + h] = oS;
    end_out  [(size_t)row * HH + h] = (_Float16)oE;
}

// ---------------------------------------------------------------------------
// Kernel 2: mid[b,o] = start_rot[b] @ weight[o]  (1024x64x64), output f16.
//   grid = (S/32, B*O), block = 256 threads (j = tid&63, 4 row-subgroups of 8).
//   weight[o] (16 KB) staged in LDS.
// ---------------------------------------------------------------------------
__global__ __launch_bounds__(256) void mid_kernel(
    const float* __restrict__ start_rot,
    const float* __restrict__ weight,
    _Float16* __restrict__ mid)
{
    const int bo  = blockIdx.y;            // b*O + o
    const int b   = bo >> 4;
    const int o   = bo & 15;
    const int j   = threadIdx.x & 63;
    const int sub = threadIdx.x >> 6;      // 0..3

    __shared__ float wl[HH * HH];
    const float* wsrc = weight + (size_t)o * HH * HH;
    for (int i = threadIdx.x; i < HH * HH; i += 256) wl[i] = wsrc[i];
    __syncthreads();

    const int row0 = blockIdx.x * 32 + sub * 8;
    for (int r = 0; r < 8; ++r) {
        const int row = row0 + r;
        const float* srow = start_rot + ((size_t)b * SS + row) * HH;
        float acc = 0.0f;
        #pragma unroll 8
        for (int i = 0; i < HH; ++i)
            acc = fmaf(srow[i], wl[i * HH + j], acc);
        mid[((size_t)bo * SS + row) * HH + j] = (_Float16)acc;
    }
}

// ---------------------------------------------------------------------------
// Kernel 3: span[b,o] = mid[b,o] @ end[b]^T  (1024x1024, K=64) with fused
// mask / tril epilogue.  One wave per 16x16 tile; block = 8 waves covering
// a 16x128 strip.  grid = (S/16, S/128, B*O).
//
// Fragment layouts (CDNA5 ISA 7.12.2, 16-bit A / B striping, wave32):
//   A 16x32: lanes 0-15 row M=lane, halves 0..7 = K 0..7,  halves 8..15 = K 16..23
//            lanes 16-31 row M=lane-16, halves = K 8..15 / K 24..31
//   B 32x16: same striping with N in place of M (B is K x N, column per lane).
//   C/D f32 16x16: VGPR r, lane l -> M = r + 8*(l>=16), N = l&15.
// ---------------------------------------------------------------------------
__global__ __launch_bounds__(256) void span_wmma_kernel(
    const _Float16* __restrict__ mid,
    const _Float16* __restrict__ endf,
    const float*    __restrict__ mask,
    float*          __restrict__ out)
{
    const int lane = threadIdx.x & 31;
    const int wave = threadIdx.x >> 5;     // 0..7
    const int x0   = blockIdx.x * 16;
    const int y0   = blockIdx.y * 128 + wave * 16;
    const int bo   = blockIdx.z;           // b*O + o
    const int b    = bo >> 4;

    const int m  = lane & 15;
    const int hi = lane >> 4;              // 0 or 1

    // A fragment: mid rows x0+m, K groups per the striping above.
    const _Float16* arow = mid  + (((size_t)bo * SS) + x0 + m) * HH + hi * 8;
    // B fragment: end^T -> B[k][n] = end[b, y0+n, k]; column n = lane&15.
    const _Float16* brow = endf + (((size_t)b  * SS) + y0 + m) * HH + hi * 8;

    union Frag { v16h v; v8h h[2]; };
    Frag a0, a1, bf0, bf1;
    a0.h[0]  = *(const v8h*)(arow +  0);   // K  0..7  / 8..15
    a0.h[1]  = *(const v8h*)(arow + 16);   // K 16..23 / 24..31
    a1.h[0]  = *(const v8h*)(arow + 32);   // K 32..39 / 40..47
    a1.h[1]  = *(const v8h*)(arow + 48);   // K 48..55 / 56..63
    bf0.h[0] = *(const v8h*)(brow +  0);
    bf0.h[1] = *(const v8h*)(brow + 16);
    bf1.h[0] = *(const v8h*)(brow + 32);
    bf1.h[1] = *(const v8h*)(brow + 48);

    v8f c = {};
    c = __builtin_amdgcn_wmma_f32_16x16x32_f16(false, a0.v, false, bf0.v,
                                               (short)0, c, false, false);
    c = __builtin_amdgcn_wmma_f32_16x16x32_f16(false, a1.v, false, bf1.v,
                                               (short)0, c, false, false);

    // Epilogue: col = y0 + (lane&15); rows rowbase..rowbase+7.
    const int   col  = y0 + m;
    const float pad  = mask[(size_t)b * SS + col];
    const float bias = -(1.0f - pad) * NEGBIG;
    const int   rowbase = x0 + hi * 8;

    #pragma unroll
    for (int r = 0; r < 8; ++r) {
        const int row = rowbase + r;
        float v = c[r] * pad + bias;
        if (col < row) v -= NEGBIG;        // strict lower triangle
        out[((size_t)bo * SS + row) * SS + col] = v;
    }
}

// ---------------------------------------------------------------------------
// Host launcher.  Inputs (setup_inputs order):
//   0 x(B,S,I) f32 | 1 mask(B,S) f32 | 2 W_start(I,H) | 3 b_start(H)
//   4 W_end(I,H)   | 5 b_end(H)      | 6 weight(O,H,H)
// Output: span (B,O,S,S) f32.
// Workspace layout: start_rot f32 (1 MB) | end_rot f16 (0.5 MB) | mid f16 (8 MB)
// ---------------------------------------------------------------------------
extern "C" void kernel_launch(void* const* d_in, const int* in_sizes, int n_in,
                              void* d_out, int out_size, void* d_ws, size_t ws_size,
                              hipStream_t stream)
{
    (void)in_sizes; (void)n_in; (void)out_size; (void)ws_size;

    const float* x   = (const float*)d_in[0];
    const float* msk = (const float*)d_in[1];
    const float* Ws  = (const float*)d_in[2];
    const float* bs  = (const float*)d_in[3];
    const float* We  = (const float*)d_in[4];
    const float* be  = (const float*)d_in[5];
    const float* wgt = (const float*)d_in[6];
    float* out = (float*)d_out;

    char* ws = (char*)d_ws;
    float*    start_rot = (float*)ws;                                  // 4096*64*4 = 1 MiB
    _Float16* end_rot   = (_Float16*)(ws + (size_t)BB * SS * HH * 4);  // 4096*64*2
    _Float16* mid       = (_Float16*)(ws + (size_t)BB * SS * HH * 4
                                         + (size_t)BB * SS * HH * 2); // 64*1024*64*2 = 8 MiB

    // K1: projections + ReLU + RoPE
    proj_rope_kernel<<<dim3(BB * SS), dim3(HH), 0, stream>>>(
        x, Ws, bs, We, be, start_rot, end_rot);

    // K2: mid[b,o] = start_rot[b] @ weight[o]
    mid_kernel<<<dim3(SS / 32, BB * OO), dim3(256), 0, stream>>>(
        start_rot, wgt, mid);

    // K3: fused WMMA GEMM + mask/tril epilogue -> 256 MB output
    span_wmma_kernel<<<dim3(SS / 16, SS / 128, BB * OO), dim3(256), 0, stream>>>(
        mid, end_rot, msk, out);
}
